// Block_8469675508074
// MI455X (gfx1250) — matrix operand order repaired and checked by reading, not compiled, not run on hardware
//
#include <hip/hip_runtime.h>
#include <math.h>

// ---------------------------------------------------------------------------
// MLA-attention + MoE transformer block for MI455X (gfx1250, wave32, WMMA).
// Heavy math runs through v_wmma_f32_16x16x32_f16 (f32 -> f16 once at LDS
// staging, f32 accumulate). Large GEMMs use 64x64 workgroup tiles with LDS
// panel reuse (8 waves, 2 WMMA accumulators each); staging is float4 loads
// + packed f16 b64 LDS stores. MoE is computed sparsely (top-2 of 16
// experts) via token gather/scatter.
// ---------------------------------------------------------------------------

typedef __attribute__((ext_vector_type(16))) _Float16 v16h;
typedef __attribute__((ext_vector_type(4)))  _Float16 h4;
typedef __attribute__((ext_vector_type(8)))  float    v8f;

constexpr int Bb = 2, Ss = 1024, Dd = 2048, Hh = 16;
constexpr int NOPE = 128, ROPE = 64, VD = 128, KVR = 512, QLR = 1536;
constexpr int NE = 16, TOPK = 2, MI = 1408, SMI = 2816;
constexpr int QKD = NOPE + ROPE;           // 192
constexpr int T = Bb * Ss;                 // 2048 tokens
constexpr int KVF = KVR + ROPE;            // 576
constexpr int CAP = T * TOPK + NE * 16;    // padded slot capacity (4352)
constexpr float ATT_SCALE = 0.07216878364870323f;  // 192^-0.5
constexpr int SROW = 1028;                 // padded LDS score row (bank spread)
constexpr int LDP  = 40;                   // padded LDS panel row (f16 elems)

#define DEVINL __device__ __forceinline__

DEVINL v8f wmma_f(v16h a, v16h b, v8f c) {
  return __builtin_amdgcn_wmma_f32_16x16x32_f16(false, a, false, b,
                                                (short)0, c, false, false);
}

// A fragment: 16x32 tile, row-major f32, origin at (row0,k0). Lane l: m=l&15,
// contiguous K run [16*half,+16).
DEVINL v16h load_a_frag(const float* A, int lda) {
  int lane = threadIdx.x & 31;
  const float* p = A + (long)(lane & 15) * lda + ((lane >> 4) * 16);
  v16h f;
#pragma unroll
  for (int i = 0; i < 8; ++i) { f[i] = (_Float16)p[i]; f[8 + i] = (_Float16)p[8 + i]; }
  return f;
}

// Same pattern, f16 source (LDS panel).
DEVINL v16h load_a_frag_h(const _Float16* A, int lda) {
  int lane = threadIdx.x & 31;
  const _Float16* p = A + (long)(lane & 15) * lda + ((lane >> 4) * 16);
  v16h f;
#pragma unroll
  for (int i = 0; i < 8; ++i) { f[i] = p[i]; f[8 + i] = p[8 + i]; }
  return f;
}

// A fragment with per-row gather (row depends only on lane&15).
DEVINL v16h load_a_frag_gather(const float* A, int lda, int row) {
  int lane = threadIdx.x & 31;
  const float* p = A + (long)row * lda + ((lane >> 4) * 16);
  v16h f;
#pragma unroll
  for (int i = 0; i < 8; ++i) { f[i] = (_Float16)p[i]; f[8 + i] = (_Float16)p[8 + i]; }
  return f;
}

// B fragment from a row-major N x K panel (B = W^T). Lane l: n = l&15,
// contiguous K run [16*half,+16) (same k order as the A loaders).
DEVINL v16h load_b_frag_nk(const float* W, int ldw) {
  int lane = threadIdx.x & 31;
  const float* p = W + (long)(lane & 15) * ldw + ((lane >> 4) * 16);
  v16h f;
#pragma unroll
  for (int i = 0; i < 16; ++i) f[i] = (_Float16)p[i];
  return f;
}

DEVINL v16h load_b_frag_nk_h(const _Float16* W, int ldw) {
  int lane = threadIdx.x & 31;
  const _Float16* p = W + (long)(lane & 15) * ldw + ((lane >> 4) * 16);
  v16h f;
#pragma unroll
  for (int i = 0; i < 16; ++i) f[i] = p[i];
  return f;
}

// B fragment from a row-major K x N matrix (strided gather along K).
DEVINL v16h load_b_frag_kn(const float* Bm, int ldb) {
  int lane = threadIdx.x & 31;
  int n = lane & 15, koff = (lane >> 4) * 16;
  v16h f;
#pragma unroll
  for (int i = 0; i < 16; ++i) f[i] = (_Float16)Bm[(long)(koff + i) * ldb + n];
  return f;
}

DEVINL h4 pack_h4(float4 v) {
  h4 r;
  r[0] = (_Float16)v.x; r[1] = (_Float16)v.y;
  r[2] = (_Float16)v.z; r[3] = (_Float16)v.w;
  return r;
}

DEVINL float silu_f(float x) { return x / (1.f + __expf(-x)); }

// ---------------------------------------------------------------------------
// RMSNorm: one block per row.
__global__ void rmsnorm_k(const float* __restrict__ x, int xs,
                          const float* __restrict__ w,
                          float* __restrict__ o, int os, int dim) {
  __shared__ float red[256];
  int row = blockIdx.x;
  const float* xr = x + (long)row * xs;
  float ss = 0.f;
  for (int i = threadIdx.x; i < dim; i += blockDim.x) { float v = xr[i]; ss += v * v; }
  red[threadIdx.x] = ss; __syncthreads();
  for (int s = blockDim.x >> 1; s > 0; s >>= 1) {
    if (threadIdx.x < (unsigned)s) red[threadIdx.x] += red[threadIdx.x + s];
    __syncthreads();
  }
  float inv = rsqrtf(red[0] / (float)dim + 1e-3f);
  float* orow = o + (long)row * os;
  for (int i = threadIdx.x; i < dim; i += blockDim.x) orow[i] = w[i] * xr[i] * inv;
}

// ---------------------------------------------------------------------------
// Blocked WMMA GEMM: C[M,N] = A[M,K] @ B + bias (+resid) (+=C if accum).
// Workgroup = 64x64 C tile, 8 waves; A/B K-panels staged in LDS as f16
// (converted once), fragments re-read from LDS. Requires M%64==0, N%64==0,
// K%32==0 (true for all call sites).
// b_kn==0: B is N x K row-major (weight). b_kn==1: B is K x N row-major.
__global__ __launch_bounds__(256) void gemm64_k(
    const float* __restrict__ A, int lda,
    const float* __restrict__ Bm, int ldb, int b_kn,
    const float* __restrict__ bias,
    const float* __restrict__ resid,
    float* __restrict__ C, int ldc,
    int M, int N, int K, int accum) {
  __shared__ _Float16 As[64 * LDP];
  __shared__ _Float16 Bs[64 * LDP];
  int tid = threadIdx.x;
  int wid = tid >> 5;
  int wm = wid & 3;        // M subtile (16 rows)
  int wn = wid >> 2;       // N subtile pair (32 cols)
  int m0 = blockIdx.y * 64, n0 = blockIdx.x * 64;
  v8f acc0 = {}, acc1 = {};

  // staging assignments: each thread owns 8 contiguous elems of one panel row
  int arow = tid >> 2, acol = (tid & 3) * 8;      // A / B(nk): 64 rows x 32 K
  const float* agp = A + (long)(m0 + arow) * lda + acol;
  _Float16* alp = As + arow * LDP + acol;
  const float* bgp = nullptr;
  _Float16* blp = Bs + arow * LDP + acol;
  int bk_kn = tid >> 3, bn_kn = (tid & 7) * 8;    // B(kn): 32 K x 64 N
  if (b_kn) bgp = Bm + (long)bk_kn * ldb + n0 + bn_kn;
  else      bgp = Bm + (long)(n0 + arow) * ldb + acol;

  for (int kb = 0; kb < K; kb += 32) {
    float4 a0 = *(const float4*)(agp);
    float4 a1 = *(const float4*)(agp + 4);
    if (b_kn) {
      const float* p = bgp + (long)kb * ldb;
      float4 b0v = *(const float4*)(p);
      float4 b1v = *(const float4*)(p + 4);
      *(h4*)(alp)     = pack_h4(a0);
      *(h4*)(alp + 4) = pack_h4(a1);
      // transpose scatter into Bs[n][k]
      Bs[(bn_kn + 0) * LDP + bk_kn] = (_Float16)b0v.x;
      Bs[(bn_kn + 1) * LDP + bk_kn] = (_Float16)b0v.y;
      Bs[(bn_kn + 2) * LDP + bk_kn] = (_Float16)b0v.z;
      Bs[(bn_kn + 3) * LDP + bk_kn] = (_Float16)b0v.w;
      Bs[(bn_kn + 4) * LDP + bk_kn] = (_Float16)b1v.x;
      Bs[(bn_kn + 5) * LDP + bk_kn] = (_Float16)b1v.y;
      Bs[(bn_kn + 6) * LDP + bk_kn] = (_Float16)b1v.z;
      Bs[(bn_kn + 7) * LDP + bk_kn] = (_Float16)b1v.w;
    } else {
      float4 b0v = *(const float4*)(bgp);
      float4 b1v = *(const float4*)(bgp + 4);
      *(h4*)(alp)     = pack_h4(a0);
      *(h4*)(alp + 4) = pack_h4(a1);
      *(h4*)(blp)     = pack_h4(b0v);
      *(h4*)(blp + 4) = pack_h4(b1v);
      if (kb + 32 < K) {  // hint next K-panel (global_prefetch_b8)
        __builtin_prefetch(agp + 32, 0, 1);
        __builtin_prefetch(bgp + 32, 0, 1);
      }
      bgp += 32;
    }
    agp += 32;
    __syncthreads();
    v16h a  = load_a_frag_h(As + wm * 16 * LDP, LDP);
    v16h b0 = load_b_frag_nk_h(Bs + (wn * 32) * LDP, LDP);
    v16h b1 = load_b_frag_nk_h(Bs + (wn * 32 + 16) * LDP, LDP);
    acc0 = wmma_f(a, b0, acc0);
    acc1 = wmma_f(a, b1, acc1);
    __syncthreads();
  }

  int lane = tid & 31, n = lane & 15, half = lane >> 4;
#pragma unroll
  for (int t2 = 0; t2 < 2; ++t2) {
    int nn = n0 + wn * 32 + t2 * 16 + n;
    float bv = bias ? bias[nn] : 0.f;
#pragma unroll
    for (int r = 0; r < 8; ++r) {
      long ci = (long)(m0 + wm * 16 + r + 8 * half) * ldc + nn;
      float v = (t2 ? acc1[r] : acc0[r]) + bv;
      if (resid) v += resid[ci];
      if (accum) C[ci] += v; else C[ci] = v;
    }
  }
}

// One-wave GEMM for the tiny gate projection (N=16).
__global__ void gemm_small_k(const float* __restrict__ A, int lda,
                             const float* __restrict__ W, int ldw,
                             const float* __restrict__ bias,
                             float* __restrict__ C, int ldc,
                             int M, int N, int K) {
  int n0 = blockIdx.x * 16, m0 = blockIdx.y * 16;
  if (m0 >= M || n0 >= N) return;
  const float* Ab = A + (long)m0 * lda;
  const float* Wb = W + (long)n0 * ldw;
  v8f acc = {};
  for (int kb = 0; kb < K; kb += 32)
    acc = wmma_f(load_a_frag(Ab + kb, lda), load_b_frag_nk(Wb + kb, ldw), acc);
  int lane = threadIdx.x & 31, n = lane & 15, half = lane >> 4;
  float bv = bias ? bias[n0 + n] : 0.f;
#pragma unroll
  for (int r = 0; r < 8; ++r)
    C[(long)(m0 + r + 8 * half) * ldc + n0 + n] = acc[r] + bv;
}

// ---------------------------------------------------------------------------
// RoPE, interleaved-pair convention. One thread per (token, head, pair).
__global__ void rope_k(float* __restrict__ x, int rowStride, int headStride,
                       int nHeads, int peOff,
                       const float* __restrict__ cs, const float* __restrict__ sn) {
  int gid = blockIdx.x * blockDim.x + threadIdx.x;
  int total = T * nHeads * (ROPE / 2);
  if (gid >= total) return;
  int i = gid & 31;                      // ROPE/2 == 32
  int h = (gid >> 5) % nHeads;
  int t = gid / (32 * nHeads);
  int s = t % Ss;
  float c = cs[s * 32 + i], sv = sn[s * 32 + i];
  float* p = x + (long)t * rowStride + h * headStride + peOff + 2 * i;
  float x0 = p[0], x1 = p[1];
  p[0] = x0 * c - x1 * sv;
  p[1] = x0 * sv + x1 * c;
}

// ---------------------------------------------------------------------------
// Attention: one workgroup (8 waves) per (b, h, 16-query tile).
// Phase 1: scores(16 x 1024) -> LDS via WMMA (nope 512 + rope 64 K-dim).
// Phase 2: row softmax (wave shuffle reductions).
// Phase 3: o_abs = P @ kv_n via WMMA, P streamed from LDS as A fragments.
__global__ void attn_k(const float* __restrict__ q,     // T x (H*QKD)
                       const float* __restrict__ qabs,  // T x H x KVR
                       const float* __restrict__ kvf,   // T x KVF (k_pe @ 512)
                       const float* __restrict__ kvn,   // T x KVR
                       float* __restrict__ oabs) {      // T x H x KVR
  extern __shared__ float sc[];   // 16 x SROW floats
  int qt = blockIdx.x, h = blockIdx.y, b = blockIdx.z;
  int wid = threadIdx.x >> 5, lane = threadIdx.x & 31;
  long tq0 = (long)b * Ss + qt * 16;
  const float* qa_org  = qabs + (tq0 * Hh + h) * KVR;
  const float* qpe_org = q + tq0 * (Hh * QKD) + h * QKD + NOPE;

  for (int kt = wid; kt < Ss / 16; kt += 8) {
    long tk0 = (long)b * Ss + kt * 16;
    const float* kn = kvn + tk0 * KVR;
    const float* kp = kvf + tk0 * KVF + KVR;
    v8f acc = {};
    for (int kb = 0; kb < KVR; kb += 32)
      acc = wmma_f(load_a_frag(qa_org + kb, Hh * KVR), load_b_frag_nk(kn + kb, KVR), acc);
    for (int kb = 0; kb < ROPE; kb += 32)
      acc = wmma_f(load_a_frag(qpe_org + kb, Hh * QKD), load_b_frag_nk(kp + kb, KVF), acc);
    int n = lane & 15, half = lane >> 4;
#pragma unroll
    for (int r = 0; r < 8; ++r)
      sc[(r + 8 * half) * SROW + kt * 16 + n] = acc[r] * ATT_SCALE;
  }
  __syncthreads();

  for (int row = wid * 2; row < wid * 2 + 2; ++row) {
    float* pr = sc + row * SROW;
    float mx = -1e30f;
    for (int c = lane; c < Ss; c += 32) mx = fmaxf(mx, pr[c]);
    for (int off = 16; off; off >>= 1) mx = fmaxf(mx, __shfl_xor(mx, off));
    float sum = 0.f;
    for (int c = lane; c < Ss; c += 32) { float e = __expf(pr[c] - mx); pr[c] = e; sum += e; }
    for (int off = 16; off; off >>= 1) sum += __shfl_xor(sum, off);
    float inv = 1.f / sum;
    for (int c = lane; c < Ss; c += 32) pr[c] *= inv;
  }
  __syncthreads();

  float* oo = oabs + (tq0 * Hh + h) * KVR;
  for (int nt = wid; nt < KVR / 16; nt += 8) {
    v8f acc = {};
    const float* vb = kvn + (long)b * Ss * KVR + nt * 16;
    for (int kb = 0; kb < Ss; kb += 32)
      acc = wmma_f(load_a_frag(sc + kb, SROW), load_b_frag_kn(vb + (long)kb * KVR, KVR), acc);
    int n = lane & 15, half = lane >> 4;
#pragma unroll
    for (int r = 0; r < 8; ++r)
      oo[(long)(r + 8 * half) * (Hh * KVR) + nt * 16 + n] = acc[r];
  }
}

// ---------------------------------------------------------------------------
// MoE routing.
__global__ void init_route_k(int* counts, int* slot_token, float* slot_scale, int cap) {
  int g = blockIdx.x * blockDim.x + threadIdx.x;
  if (g < NE) counts[g] = 0;
  if (g < cap) { slot_token[g] = 0; slot_scale[g] = 0.f; }
}

// One wave per token: softmax over 16 logits, top-2 of (prob + gate_bias).
__global__ void route_k(const float* __restrict__ logits, const float* __restrict__ gbias,
                        int* __restrict__ eidx, float* __restrict__ ewt,
                        int* __restrict__ counts, int* __restrict__ slotpos) {
  int t = blockIdx.x, lane = threadIdx.x;
  float lg = (lane < NE) ? logits[t * NE + lane] : -1e30f;
  float mx = lg;
  for (int off = 16; off; off >>= 1) mx = fmaxf(mx, __shfl_xor(mx, off));
  float e = (lane < NE) ? __expf(lg - mx) : 0.f;
  float sum = e;
  for (int off = 16; off; off >>= 1) sum += __shfl_xor(sum, off);
  float prob = e / sum;
  float key = (lane < NE) ? prob + gbias[lane] : -1e30f;
  for (int k = 0; k < TOPK; ++k) {
    float bk = key; int bi = lane;
    for (int off = 16; off; off >>= 1) {
      float ok = __shfl_xor(bk, off); int oi = __shfl_xor(bi, off);
      if (ok > bk || (ok == bk && oi < bi)) { bk = ok; bi = oi; }
    }
    float w = __shfl(prob, bi);
    if (lane == 0) {
      eidx[t * TOPK + k] = bi;
      ewt[t * TOPK + k] = w;
      slotpos[t * TOPK + k] = atomicAdd(&counts[bi], 1);
    }
    if (lane == bi) key = -1e30f;
  }
}

__global__ void scan_k(const int* __restrict__ counts, int* __restrict__ base) {
  if (threadIdx.x == 0 && blockIdx.x == 0) {
    int acc = 0;
    for (int e = 0; e < NE; ++e) { base[e] = acc; acc += (counts[e] + 15) & ~15; }
    base[NE] = acc;
  }
}

__global__ void fill_slots_k(const int* __restrict__ eidx, const float* __restrict__ ewt,
                             const int* __restrict__ slotpos, const int* __restrict__ base,
                             int* __restrict__ slot_token, float* __restrict__ slot_scale) {
  int g = blockIdx.x * blockDim.x + threadIdx.x;
  if (g >= T * TOPK) return;
  int e = eidx[g];
  int slot = base[e] + slotpos[g];
  slot_token[slot] = g / TOPK;
  slot_scale[slot] = ewt[g];
}

// out += h2 + combine @ e_b2   (residual + MoE output bias, elementwise)
__global__ void add_h2_ebias_k(float* __restrict__ out, const float* __restrict__ h2,
                               const float* __restrict__ e_b2,
                               const int* __restrict__ eidx, const float* __restrict__ ewt) {
  long g = (long)blockIdx.x * blockDim.x + threadIdx.x;
  if (g >= (long)T * Dd) return;
  int t = (int)(g / Dd), d = (int)(g % Dd);
  int i0 = eidx[t * 2], i1 = eidx[t * 2 + 1];
  out[g] += h2[g] + ewt[t * 2] * e_b2[(long)i0 * Dd + d]
                  + ewt[t * 2 + 1] * e_b2[(long)i1 * Dd + d];
}

// Per-expert gathered dual GEMM: act = silu(h2@w1^T+b1) * (h2@w3^T+b3) * wt.
// grid = (MI/16, T/16 worst-case tiles, NE); early-exit past padded count.
__global__ void expert_act_k(const float* __restrict__ h2,
                             const float* __restrict__ w1, const float* __restrict__ b1,
                             const float* __restrict__ w3, const float* __restrict__ b3,
                             const int* __restrict__ base, const int* __restrict__ slot_token,
                             const float* __restrict__ slot_scale,
                             float* __restrict__ act) {
  int e = blockIdx.z, m0 = blockIdx.y * 16, n0 = blockIdx.x * 16;
  int cnt = base[e + 1] - base[e];          // padded count
  if (m0 >= cnt) return;
  int lane = threadIdx.x & 31;
  int tok = slot_token[base[e] + m0 + (lane & 15)];
  const float* W1 = w1 + (long)e * MI * Dd + (long)n0 * Dd;
  const float* W3 = w3 + (long)e * MI * Dd + (long)n0 * Dd;
  v8f a1 = {}, a3 = {};
  for (int kb = 0; kb < Dd; kb += 32) {
    v16h a = load_a_frag_gather(h2 + kb, Dd, tok);
    a1 = wmma_f(a, load_b_frag_nk(W1 + kb, Dd), a1);
    a3 = wmma_f(a, load_b_frag_nk(W3 + kb, Dd), a3);
  }
  int n = lane & 15, half = lane >> 4;
  float bb1 = b1[e * MI + n0 + n], bb3 = b3[e * MI + n0 + n];
#pragma unroll
  for (int r = 0; r < 8; ++r) {
    int s = base[e] + m0 + r + 8 * half;
    float v1 = a1[r] + bb1, v3 = a3[r] + bb3;
    act[(long)s * MI + n0 + n] = silu_f(v1) * v3 * slot_scale[s];
  }
}

// Per-expert down-projection + scatter-add into out (atomic f32 adds).
__global__ void expert_down_k(const float* __restrict__ act, const float* __restrict__ w2,
                              const int* __restrict__ base, const int* __restrict__ slot_token,
                              float* __restrict__ out) {
  int e = blockIdx.z, m0 = blockIdx.y * 16, n0 = blockIdx.x * 16;
  int cnt = base[e + 1] - base[e];
  if (m0 >= cnt) return;
  const float* Ab = act + (long)(base[e] + m0) * MI;
  const float* W = w2 + (long)e * Dd * MI + (long)n0 * MI;
  v8f acc = {};
  for (int kb = 0; kb < MI; kb += 32)
    acc = wmma_f(load_a_frag(Ab + kb, MI), load_b_frag_nk(W + kb, MI), acc);
  int lane = threadIdx.x & 31, n = lane & 15, half = lane >> 4;
#pragma unroll
  for (int r = 0; r < 8; ++r) {
    int s = base[e] + m0 + r + 8 * half;
    int tok = slot_token[s];
    atomicAdd(&out[(long)tok * Dd + n0 + n], acc[r]);
  }
}

__global__ void silu_mul_k(const float* __restrict__ a, const float* __restrict__ g3,
                           float* __restrict__ o, long n) {
  long i = (long)blockIdx.x * blockDim.x + threadIdx.x;
  if (i < n) o[i] = silu_f(a[i]) * g3[i];
}

// ---------------------------------------------------------------------------
extern "C" void kernel_launch(void* const* d_in, const int* in_sizes, int n_in,
                              void* d_out, int out_size, void* d_ws, size_t ws_size,
                              hipStream_t stream) {
  (void)in_sizes; (void)n_in; (void)out_size; (void)ws_size;
  const float* x          = (const float*)d_in[0];
  const float* attn_nw    = (const float*)d_in[1];
  const float* ffn_nw     = (const float*)d_in[2];
  const float* wq_a_w     = (const float*)d_in[3];
  const float* wq_a_b     = (const float*)d_in[4];
  const float* q_norm_w   = (const float*)d_in[5];
  const float* wq_b_w     = (const float*)d_in[6];
  const float* wq_b_b     = (const float*)d_in[7];
  const float* wkv_a_w    = (const float*)d_in[8];
  const float* wkv_a_b    = (const float*)d_in[9];
  const float* kv_norm_w  = (const float*)d_in[10];
  const float* wkv_b_w    = (const float*)d_in[11];
  const float* wo_w       = (const float*)d_in[12];
  const float* wo_b       = (const float*)d_in[13];
  const float* gate_w     = (const float*)d_in[14];
  const float* gate_b     = (const float*)d_in[15];
  const float* e_w1       = (const float*)d_in[16];
  const float* e_b1       = (const float*)d_in[17];
  const float* e_w2       = (const float*)d_in[18];
  const float* e_b2       = (const float*)d_in[19];
  const float* e_w3       = (const float*)d_in[20];
  const float* e_b3       = (const float*)d_in[21];
  const float* s_w1       = (const float*)d_in[22];
  const float* s_b1       = (const float*)d_in[23];
  const float* s_w2       = (const float*)d_in[24];
  const float* s_b2       = (const float*)d_in[25];
  const float* s_w3       = (const float*)d_in[26];
  const float* s_b3       = (const float*)d_in[27];
  const float* cosb       = (const float*)d_in[28];
  const float* sinb       = (const float*)d_in[29];
  float* out = (float*)d_out;

  // ---- workspace bump allocator (f32 region then int region) ----
  float* wsf = (float*)d_ws;
  size_t off = 0;
  auto alloc = [&](size_t n) { float* p = wsf + off; off += n; return p; };
  float* h     = alloc((size_t)T * Dd);
  float* qlat  = alloc((size_t)T * QLR);
  float* qlatn = alloc((size_t)T * QLR);
  float* q     = alloc((size_t)T * Hh * QKD);
  float* kvf   = alloc((size_t)T * KVF);
  float* kvn   = alloc((size_t)T * KVR);
  float* qabs  = alloc((size_t)T * Hh * KVR);
  float* oabs  = alloc((size_t)T * Hh * KVR);
  float* oproj = alloc((size_t)T * Hh * VD);
  float* h2    = alloc((size_t)T * Dd);
  float* glog  = alloc((size_t)T * NE);
  float* act   = alloc((size_t)CAP * MI);
  float* z1    = alloc((size_t)T * SMI);
  float* z3    = alloc((size_t)T * SMI);
  float* zact  = alloc((size_t)T * SMI);
  float* ewt   = alloc((size_t)T * TOPK);
  float* sscal = alloc((size_t)CAP);
  int* ip = (int*)(wsf + off);
  int* counts  = ip; ip += NE;
  int* ebase   = ip; ip += NE + 1;
  int* eidx    = ip; ip += T * TOPK;
  int* slotpos = ip; ip += T * TOPK;
  int* stok    = ip; ip += CAP;

  dim3 blk32(32), blk256(256);

  // ---- attention path ----
  rmsnorm_k<<<T, 256, 0, stream>>>(x, Dd, attn_nw, h, Dd, Dd);
  gemm64_k<<<dim3(QLR / 64, T / 64), blk256, 0, stream>>>(
      h, Dd, wq_a_w, Dd, 0, wq_a_b, nullptr, qlat, QLR, T, QLR, Dd, 0);
  rmsnorm_k<<<T, 256, 0, stream>>>(qlat, QLR, q_norm_w, qlatn, QLR, QLR);
  gemm64_k<<<dim3(Hh * QKD / 64, T / 64), blk256, 0, stream>>>(
      qlatn, QLR, wq_b_w, QLR, 0, wq_b_b, nullptr, q, Hh * QKD, T, Hh * QKD, QLR, 0);
  gemm64_k<<<dim3(KVF / 64, T / 64), blk256, 0, stream>>>(
      h, Dd, wkv_a_w, Dd, 0, wkv_a_b, nullptr, kvf, KVF, T, KVF, Dd, 0);

  rope_k<<<(T * Hh * 32 + 255) / 256, 256, 0, stream>>>(q, Hh * QKD, QKD, Hh, NOPE, cosb, sinb);
  rope_k<<<(T * 32 + 255) / 256, 256, 0, stream>>>(kvf, KVF, 0, 1, KVR, cosb, sinb);

  rmsnorm_k<<<T, 256, 0, stream>>>(kvf, KVF, kv_norm_w, kvn, KVR, KVR);

  // q_abs[t,h,:] = q_nope[t,h,:] @ wb[h,:NOPE,:]   (weight is K x N here)
  for (int hh = 0; hh < Hh; ++hh)
    gemm64_k<<<dim3(KVR / 64, T / 64), blk256, 0, stream>>>(
        q + hh * QKD, Hh * QKD,
        wkv_b_w + (long)hh * (NOPE + VD) * KVR, KVR, 1,
        nullptr, nullptr, qabs + hh * KVR, Hh * KVR, T, KVR, NOPE, 0);

  attn_k<<<dim3(Ss / 16, Hh, Bb), 256, 16 * SROW * sizeof(float), stream>>>(
      q, qabs, kvf, kvn, oabs);

  // o_proj[t,h,:] = o_abs[t,h,:] @ wb[h,NOPE:,:]^T  (weight is N x K here)
  for (int hh = 0; hh < Hh; ++hh)
    gemm64_k<<<dim3(VD / 64, T / 64), blk256, 0, stream>>>(
        oabs + hh * KVR, Hh * KVR,
        wkv_b_w + ((long)hh * (NOPE + VD) + NOPE) * KVR, KVR, 0,
        nullptr, nullptr, oproj + hh * VD, Hh * VD, T, VD, KVR, 0);

  // out = x + o_proj @ wo^T + wo_b
  gemm64_k<<<dim3(Dd / 64, T / 64), blk256, 0, stream>>>(
      oproj, Hh * VD, wo_w, Hh * VD, 0, wo_b, x, out, Dd, T, Dd, Hh * VD, 0);

  // ---- FFN path ----
  rmsnorm_k<<<T, 256, 0, stream>>>(out, Dd, ffn_nw, h2, Dd, Dd);

  // gate + routing
  gemm_small_k<<<dim3(NE / 16, T / 16), blk32, 0, stream>>>(
      h2, Dd, gate_w, Dd, gate_b, glog, NE, T, NE, Dd);
  init_route_k<<<(CAP + 255) / 256, 256, 0, stream>>>(counts, stok, sscal, CAP);
  route_k<<<T, 32, 0, stream>>>(glog, gate_b, eidx, ewt, counts, slotpos);
  scan_k<<<1, 32, 0, stream>>>(counts, ebase);
  fill_slots_k<<<(T * TOPK + 255) / 256, 256, 0, stream>>>(eidx, ewt, slotpos, ebase, stok, sscal);

  // residual h2 + MoE bias term
  add_h2_ebias_k<<<(int)(((long)T * Dd + 255) / 256), 256, 0, stream>>>(out, h2, e_b2, eidx, ewt);

  // sparse expert GEMMs (token-gathered, worst-case grid + early exit)
  expert_act_k<<<dim3(MI / 16, T / 16, NE), blk32, 0, stream>>>(
      h2, e_w1, e_b1, e_w3, e_b3, ebase, stok, sscal, act);
  expert_down_k<<<dim3(Dd / 16, T / 16, NE), blk32, 0, stream>>>(
      act, e_w2, ebase, stok, out);

  // shared expert
  gemm64_k<<<dim3(SMI / 64, T / 64), blk256, 0, stream>>>(
      h2, Dd, s_w1, Dd, 0, s_b1, nullptr, z1, SMI, T, SMI, Dd, 0);
  gemm64_k<<<dim3(SMI / 64, T / 64), blk256, 0, stream>>>(
      h2, Dd, s_w3, Dd, 0, s_b3, nullptr, z3, SMI, T, SMI, Dd, 0);
  silu_mul_k<<<(int)(((long)T * SMI + 255) / 256), 256, 0, stream>>>(z1, z3, zact, (long)T * SMI);
  gemm64_k<<<dim3(Dd / 64, T / 64), blk256, 0, stream>>>(
      zact, SMI, s_w2, SMI, 0, s_b2, nullptr, out, Dd, T, Dd, SMI, 1);
}
